// Block_1855425872448
// MI455X (gfx1250) — compile-verified, matrix operand order checked by reading, not run on hardware
//
#include <hip/hip_runtime.h>
#include <hip/hip_bf16.h>
#include <math.h>

// ---------------------------------------------------------------------------
// Problem constants (fixed by the reference)
// ---------------------------------------------------------------------------
constexpr int Bb  = 2;
constexpr int Nn  = 2048;
constexpr int Cc  = 1024;
constexpr int Hh  = 16;
constexpr int HDd = 64;     // head dim
constexpr int HID = 4096;
constexpr int Mrows = Bb * Nn;   // 4096 token rows

typedef __bf16 bf16_t;
typedef __bf16 v16bf __attribute__((ext_vector_type(16)));
typedef float  v8f   __attribute__((ext_vector_type(8)));
typedef int    v4i   __attribute__((vector_size(16)));

#define WMMA_BF16(a, b, c) \
    __builtin_amdgcn_wmma_f32_16x16x32_bf16(false, (a), false, (b), (short)0, (c), false, false)

// ---------------------------------------------------------------------------
// Async global->LDS copy (CDNA5 GLOBAL_LOAD_ASYNC_TO_LDS_B128, ASYNCcnt), with
// a synchronous float4 fallback if the builtin is unavailable.
// Builtin signature (from hipcc diagnostic): (v4i AS1*, v4i AS3*, int, int)
// ---------------------------------------------------------------------------
#if defined(__has_builtin)
#  if __has_builtin(__builtin_amdgcn_global_load_async_to_lds_b128)
#    define HAVE_ASYNC_LDS 1
#  endif
#endif
#ifndef HAVE_ASYNC_LDS
#  define HAVE_ASYNC_LDS 0
#endif

__device__ __forceinline__ void async_copy16(const bf16_t* g, bf16_t* l) {
#if HAVE_ASYNC_LDS
    __builtin_amdgcn_global_load_async_to_lds_b128(
        (__attribute__((address_space(1))) v4i*)g,
        (__attribute__((address_space(3))) v4i*)l, 0, 0);
#else
    *(float4*)l = *(const float4*)g;
#endif
}

__device__ __forceinline__ void wait_async_le4() {
#if HAVE_ASYNC_LDS
    asm volatile("s_wait_asynccnt 0x4" ::: "memory");
#endif
}
__device__ __forceinline__ void wait_async_0() {
#if HAVE_ASYNC_LDS
    asm volatile("s_wait_asynccnt 0x0" ::: "memory");
#endif
}

// ---------------------------------------------------------------------------
// f32 -> bf16 elementwise convert (weights)
// ---------------------------------------------------------------------------
__global__ void f32_to_bf16_kernel(const float* __restrict__ in,
                                   bf16_t* __restrict__ out, int n) {
    int i = blockIdx.x * blockDim.x + threadIdx.x;
    int stride = gridDim.x * blockDim.x;
    for (; i < n; i += stride) out[i] = (bf16_t)in[i];
}

// qkv_bias = concat(q_bias, zeros, v_bias)  (3C floats)
__global__ void build_qkv_bias_kernel(const float* __restrict__ qb,
                                      const float* __restrict__ vb,
                                      float* __restrict__ out) {
    int i = blockIdx.x * blockDim.x + threadIdx.x;
    if (i < Cc)            out[i] = qb[i];
    else if (i < 2 * Cc)   out[i] = 0.0f;
    else if (i < 3 * Cc)   out[i] = vb[i - 2 * Cc];
}

// ---------------------------------------------------------------------------
// LayerNorm over last dim (C=1024), output bf16. One block (256 thr) per row.
// ---------------------------------------------------------------------------
__global__ void layernorm_bf16_kernel(const float* __restrict__ x,
                                      const float* __restrict__ g,
                                      const float* __restrict__ b,
                                      bf16_t* __restrict__ out, int C) {
    const int row = blockIdx.x;
    const float* xr = x + (size_t)row * C;

    float s = 0.0f, ss = 0.0f;
    for (int i = threadIdx.x; i < C; i += blockDim.x) {
        float v = xr[i];
        s += v; ss += v * v;
    }
    for (int off = 16; off; off >>= 1) {
        s  += __shfl_xor(s,  off, 32);
        ss += __shfl_xor(ss, off, 32);
    }
    __shared__ float sh[16];
    const int wv = threadIdx.x >> 5, ln = threadIdx.x & 31;
    if (ln == 0) { sh[wv] = s; sh[8 + wv] = ss; }
    __syncthreads();
    if (threadIdx.x == 0) {
        float ts = 0.0f, tss = 0.0f;
        for (int w = 0; w < 8; ++w) { ts += sh[w]; tss += sh[8 + w]; }
        sh[0] = ts; sh[8] = tss;
    }
    __syncthreads();
    const float mu   = sh[0] / (float)C;
    const float var  = sh[8] / (float)C - mu * mu;
    const float rinv = rsqrtf(var + 1e-5f);

    bf16_t* orow = out + (size_t)row * C;
    for (int i = threadIdx.x; i < C; i += blockDim.x)
        orow[i] = (bf16_t)((xr[i] - mu) * rinv * g[i] + b[i]);
}

// ---------------------------------------------------------------------------
// Tiled bf16 WMMA GEMM:  D[M,N] = A[M,K] @ W[N,K]^T  (+ epilogue)
//   Block tile 128x128, 256 threads = 8 waves, wave tile 64(M) x 32(N),
//   K step 32, DOUBLE-BUFFERED async global->LDS staging.
// EPI: 0 = +bias, store bf16           (QKV)
//      1 = +bias +residual, store f32  (PROJ / FC2)
//      2 = +bias, store f32 (h) AND exact GELU -> bf16 (FC1)
// ---------------------------------------------------------------------------
template <int EPI>
__global__ __launch_bounds__(256)
void gemm_bf16_kernel(const bf16_t* __restrict__ A, const bf16_t* __restrict__ W,
                      const float* __restrict__ bias, const float* __restrict__ res,
                      float* __restrict__ outF, bf16_t* __restrict__ outB,
                      int M, int N, int K) {
    __shared__ bf16_t As[2][128 * 32];
    __shared__ bf16_t Bs[2][128 * 32];

    const int tid  = threadIdx.x;
    const int lane = tid & 31;
    const int wv   = tid >> 5;
    const int wrow = wv >> 2;        // 0..1 -> 64 rows each
    const int wcol = wv & 3;         // 0..3 -> 32 cols each
    const int lrow = lane & 15;
    const int hv   = lane >> 4;      // half of the wave
    const int bm   = blockIdx.y * 128;
    const int bn   = blockIdx.x * 128;

    // this thread's two 16B chunks per 128x32 tile (512 chunks / 256 threads)
    const int c0 = tid, c1 = tid + 256;
    const int r0 = c0 >> 2, ch0 = (c0 & 3) * 8;
    const int r1 = c1 >> 2, ch1 = (c1 & 3) * 8;

    auto issue_tile = [&](int buf, int k0) {
        async_copy16(A + (size_t)(bm + r0) * K + k0 + ch0, &As[buf][r0 * 32 + ch0]);
        async_copy16(W + (size_t)(bn + r0) * K + k0 + ch0, &Bs[buf][r0 * 32 + ch0]);
        async_copy16(A + (size_t)(bm + r1) * K + k0 + ch1, &As[buf][r1 * 32 + ch1]);
        async_copy16(W + (size_t)(bn + r1) * K + k0 + ch1, &Bs[buf][r1 * 32 + ch1]);
    };

    v8f acc[4][2];
#pragma unroll
    for (int mi = 0; mi < 4; ++mi)
#pragma unroll
        for (int ni = 0; ni < 2; ++ni) acc[mi][ni] = (v8f){};

    const int nk = K >> 5;
    issue_tile(0, 0);
    for (int t = 0; t < nk; ++t) {
        const int cur = t & 1;
        const bool more = (t + 1) < nk;
        if (more) issue_tile(cur ^ 1, (t + 1) << 5);
        if (more) wait_async_le4(); else wait_async_0();
        __syncthreads();

        v16bf af[4], bf[2];
#pragma unroll
        for (int mi = 0; mi < 4; ++mi)
            af[mi] = *(const v16bf*)(&As[cur][(wrow * 64 + mi * 16 + lrow) * 32 + hv * 16]);
#pragma unroll
        for (int ni = 0; ni < 2; ++ni)
            bf[ni] = *(const v16bf*)(&Bs[cur][(wcol * 32 + ni * 16 + lrow) * 32 + hv * 16]);

#pragma unroll
        for (int mi = 0; mi < 4; ++mi)
#pragma unroll
            for (int ni = 0; ni < 2; ++ni)
                acc[mi][ni] = WMMA_BF16(af[mi], bf[ni], acc[mi][ni]);
        __syncthreads();
    }

    // epilogue (C layout: lane -> col = lrow, rows = hv*8 + i)
#pragma unroll
    for (int mi = 0; mi < 4; ++mi) {
#pragma unroll
        for (int ni = 0; ni < 2; ++ni) {
            const int gcol = bn + wcol * 32 + ni * 16 + lrow;
            const float bc = bias[gcol];
#pragma unroll
            for (int i = 0; i < 8; ++i) {
                const int grow = bm + wrow * 64 + mi * 16 + hv * 8 + i;
                const size_t o = (size_t)grow * N + gcol;
                const float v = acc[mi][ni][i] + bc;
                if constexpr (EPI == 0) {
                    outB[o] = (bf16_t)v;
                } else if constexpr (EPI == 1) {
                    outF[o] = v + res[o];
                } else {
                    outF[o] = v;                                   // x_interm
                    const float gel = 0.5f * v * (1.0f + erff(v * 0.70710678118f));
                    outB[o] = (bf16_t)gel;
                }
            }
        }
    }
}

// ---------------------------------------------------------------------------
// Flash attention (bf16 WMMA, f32 online softmax).
//   qkv: bf16 [B*N, 3C] rows; q at col 0, k at col C, v at col 2C, head h at +h*64
//   Each wave owns 16 queries; block = 8 waves = 128 queries for one (b,h).
//   Key loop: 32 keys/iter; K tile async-staged, V staged transposed.
//   Row sums computed on the matrix unit: rowsum(P) = P @ ones.
// ---------------------------------------------------------------------------
__global__ __launch_bounds__(256)
void flash_attn_kernel(const bf16_t* __restrict__ qkv,
                       const unsigned char* __restrict__ mask,
                       bf16_t* __restrict__ out) {
    __shared__ bf16_t Kt[32 * 64];      // [key][hd]
    __shared__ bf16_t Vt[64 * 32];      // [hd][key]  (transposed)
    __shared__ bf16_t Pw[8][16 * 32];   // per-wave P tile
    __shared__ unsigned char msk[32];

    const int tid  = threadIdx.x;
    const int lane = tid & 31;
    const int wv   = tid >> 5;
    const int lrow = lane & 15;
    const int hv   = lane >> 4;
    const int h    = blockIdx.y;
    const int b    = blockIdx.z;
    const int q0   = blockIdx.x * 128 + wv * 16;
    const size_t rstride = 3 * Cc;

    // Q fragment (16 queries x 64), two K=32 halves, loaded once
    const bf16_t* qbase = qkv + ((size_t)(b * Nn) + q0) * rstride + h * HDd;
    v16bf qf0 = *(const v16bf*)(qbase + (size_t)lrow * rstride + hv * 16);
    v16bf qf1 = *(const v16bf*)(qbase + (size_t)lrow * rstride + 32 + hv * 16);

    // all-ones B fragment: rowsum(P) = P @ ones  (one extra WMMA per iter)
    v16bf ones;
#pragma unroll
    for (int e = 0; e < 16; ++e) ones[e] = (bf16_t)1.0f;

    v8f o[4];
#pragma unroll
    for (int nf = 0; nf < 4; ++nf) o[nf] = (v8f){};
    float mrow[8], lsum[8];
#pragma unroll
    for (int i = 0; i < 8; ++i) { mrow[i] = -INFINITY; lsum[i] = 0.0f; }

    const float scale = 0.125f;   // hd^-0.5 = 1/8
    const int r  = tid >> 3;      // key row 0..31 for cooperative staging
    const int ch = tid & 7;       // 8-elem chunk of hd

    for (int kb = 0; kb < Nn; kb += 32) {
        // ---- cooperative staging: K async (no transpose), V via VGPR transpose
        {
            const bf16_t* kro = qkv + ((size_t)(b * Nn) + kb + r) * rstride
                                + Cc + h * HDd + ch * 8;
            async_copy16(kro, &Kt[r * 64 + ch * 8]);
            bf16_t tmp[8];
            *(float4*)tmp = *(const float4*)(kro + Cc);   // V row chunk
#pragma unroll
            for (int e = 0; e < 8; ++e) Vt[(ch * 8 + e) * 32 + r] = tmp[e];
            if (tid < 32) msk[tid] = mask[(size_t)b * Nn + kb + tid];
        }
        wait_async_0();
        __syncthreads();

        // ---- S = (Q * scale) @ K^T  (two 16-key subtiles) ----
        v8f s[2];
#pragma unroll
        for (int sub = 0; sub < 2; ++sub) {
            v16bf kf0 = *(const v16bf*)(&Kt[(sub * 16 + lrow) * 64 + hv * 16]);
            v16bf kf1 = *(const v16bf*)(&Kt[(sub * 16 + lrow) * 64 + 32 + hv * 16]);
            v8f sv = (v8f){};
            sv = WMMA_BF16(qf0, kf0, sv);
            sv = WMMA_BF16(qf1, kf1, sv);
            s[sub] = sv;
        }
#pragma unroll
        for (int sub = 0; sub < 2; ++sub) {
            const bool mk = msk[sub * 16 + lrow] != 0;
#pragma unroll
            for (int i = 0; i < 8; ++i)
                s[sub][i] = mk ? -INFINITY : s[sub][i] * scale;
        }

        // ---- online softmax: row max via lane butterfly, then P = exp ----
        float alpha[8];
#pragma unroll
        for (int i = 0; i < 8; ++i) {
            float mx = fmaxf(s[0][i], s[1][i]);
            for (int off = 8; off; off >>= 1) mx = fmaxf(mx, __shfl_xor(mx, off, 32));
            const float mnew = fmaxf(mrow[i], mx);
            alpha[i] = __expf(mrow[i] - mnew);
            mrow[i] = mnew;
            s[0][i] = __expf(s[0][i] - mnew);
            s[1][i] = __expf(s[1][i] - mnew);
        }
#pragma unroll
        for (int nf = 0; nf < 4; ++nf)
#pragma unroll
            for (int i = 0; i < 8; ++i) o[nf][i] *= alpha[i];

        // ---- P (C-layout f32) -> per-wave LDS -> A-layout bf16 fragment ----
#pragma unroll
        for (int sub = 0; sub < 2; ++sub)
#pragma unroll
            for (int i = 0; i < 8; ++i)
                Pw[wv][(hv * 8 + i) * 32 + sub * 16 + lrow] = (bf16_t)s[sub][i];
        asm volatile("s_wait_dscnt 0" ::: "memory");   // wave-local LDS RAW
        v16bf pa = *(const v16bf*)(&Pw[wv][lrow * 32 + hv * 16]);

        // ---- row sums on the matrix unit: lsum = lsum*alpha + (P @ ones) ----
        v8f ls = WMMA_BF16(pa, ones, (v8f){});
#pragma unroll
        for (int i = 0; i < 8; ++i) lsum[i] = lsum[i] * alpha[i] + ls[i];

        // ---- O += P @ V  (4 hd-column subtiles) ----
#pragma unroll
        for (int nf = 0; nf < 4; ++nf) {
            v16bf vb = *(const v16bf*)(&Vt[(nf * 16 + lrow) * 32 + hv * 16]);
            o[nf] = WMMA_BF16(pa, vb, o[nf]);
        }
        __syncthreads();
    }

    // ---- finalize: O / l, store bf16 into [B*N, C] at head column ----
    bf16_t* obase = out + ((size_t)(b * Nn) + q0) * Cc + h * HDd;
#pragma unroll
    for (int i = 0; i < 8; ++i) {
        const float inv = lsum[i] > 0.0f ? 1.0f / lsum[i] : 0.0f;
        const int rr = hv * 8 + i;
#pragma unroll
        for (int nf = 0; nf < 4; ++nf)
            obase[(size_t)rr * Cc + nf * 16 + lrow] = (bf16_t)(o[nf][i] * inv);
    }
}

// ---------------------------------------------------------------------------
// Host-side orchestration
// ---------------------------------------------------------------------------
extern "C" void kernel_launch(void* const* d_in, const int* in_sizes, int n_in,
                              void* d_out, int out_size, void* d_ws, size_t ws_size,
                              hipStream_t stream) {
    (void)in_sizes; (void)n_in; (void)out_size; (void)ws_size;

    const float* x      = (const float*)d_in[0];
    const unsigned char* mask = (const unsigned char*)d_in[1];   // bool
    const float* ln1_g  = (const float*)d_in[2];
    const float* ln1_b  = (const float*)d_in[3];
    const float* qkv_w  = (const float*)d_in[4];
    const float* q_bias = (const float*)d_in[5];
    const float* v_bias = (const float*)d_in[6];
    const float* proj_w = (const float*)d_in[7];
    const float* proj_b = (const float*)d_in[8];
    const float* ln2_g  = (const float*)d_in[9];
    const float* ln2_b  = (const float*)d_in[10];
    const float* fc1_w  = (const float*)d_in[11];
    const float* fc1_b  = (const float*)d_in[12];
    const float* fc2_w  = (const float*)d_in[13];
    const float* fc2_b  = (const float*)d_in[14];

    float* out_h = (float*)d_out;                                  // x_interm [M,HID]
    float* out_x = (float*)d_out + (size_t)Mrows * HID;            // final x  [M,C]

    // carve workspace
    char* w = (char*)d_ws;
    auto carve = [&](size_t bytes) {
        void* p = (void*)w;
        w += (bytes + 255) & ~(size_t)255;
        return p;
    };
    bf16_t* xn_bf    = (bf16_t*)carve((size_t)Mrows * Cc * 2);
    bf16_t* wqkv_bf  = (bf16_t*)carve((size_t)3 * Cc * Cc * 2);
    bf16_t* wproj_bf = (bf16_t*)carve((size_t)Cc * Cc * 2);
    bf16_t* wfc1_bf  = (bf16_t*)carve((size_t)HID * Cc * 2);
    bf16_t* wfc2_bf  = (bf16_t*)carve((size_t)Cc * HID * 2);
    bf16_t* qkv_bf   = (bf16_t*)carve((size_t)Mrows * 3 * Cc * 2);
    bf16_t* attn_bf  = (bf16_t*)carve((size_t)Mrows * Cc * 2);
    float*  x1       = (float*) carve((size_t)Mrows * Cc * 4);
    bf16_t* x2n_bf   = (bf16_t*)carve((size_t)Mrows * Cc * 2);
    bf16_t* hact_bf  = (bf16_t*)carve((size_t)Mrows * HID * 2);
    float*  qkvbias  = (float*) carve((size_t)3 * Cc * 4);

    // 1) weight converts + fused qkv bias
    f32_to_bf16_kernel<<<1024, 256, 0, stream>>>(qkv_w, wqkv_bf, 3 * Cc * Cc);
    f32_to_bf16_kernel<<<1024, 256, 0, stream>>>(proj_w, wproj_bf, Cc * Cc);
    f32_to_bf16_kernel<<<1024, 256, 0, stream>>>(fc1_w, wfc1_bf, HID * Cc);
    f32_to_bf16_kernel<<<1024, 256, 0, stream>>>(fc2_w, wfc2_bf, Cc * HID);
    build_qkv_bias_kernel<<<12, 256, 0, stream>>>(q_bias, v_bias, qkvbias);

    // 2) LN1 -> bf16
    layernorm_bf16_kernel<<<Mrows, 256, 0, stream>>>(x, ln1_g, ln1_b, xn_bf, Cc);

    // 3) QKV GEMM: [4096,1024] @ [3072,1024]^T -> bf16 [4096,3072] (+bias)
    gemm_bf16_kernel<0><<<dim3(3 * Cc / 128, Mrows / 128), 256, 0, stream>>>(
        xn_bf, wqkv_bf, qkvbias, nullptr, nullptr, qkv_bf, Mrows, 3 * Cc, Cc);

    // 4) flash attention -> bf16 [4096,1024]
    flash_attn_kernel<<<dim3(Nn / 128, Hh, Bb), 256, 0, stream>>>(qkv_bf, mask, attn_bf);

    // 5) proj GEMM + residual: x1 = attn @ proj_w^T + proj_b + x
    gemm_bf16_kernel<1><<<dim3(Cc / 128, Mrows / 128), 256, 0, stream>>>(
        attn_bf, wproj_bf, proj_b, x, x1, nullptr, Mrows, Cc, Cc);

    // 6) LN2 -> bf16
    layernorm_bf16_kernel<<<Mrows, 256, 0, stream>>>(x1, ln2_g, ln2_b, x2n_bf, Cc);

    // 7) FC1: h = x2n @ fc1_w^T + fc1_b -> out_h (f32)  AND gelu(h) -> bf16
    gemm_bf16_kernel<2><<<dim3(HID / 128, Mrows / 128), 256, 0, stream>>>(
        x2n_bf, wfc1_bf, fc1_b, nullptr, out_h, hact_bf, Mrows, HID, Cc);

    // 8) FC2 + residual: out_x = gelu(h) @ fc2_w^T + fc2_b + x1
    gemm_bf16_kernel<1><<<dim3(Cc / 128, Mrows / 128), 256, 0, stream>>>(
        hact_bf, wfc2_bf, fc2_b, x1, out_x, nullptr, Mrows, Cc, HID);
}
